// DeformableTransformer_33036888440869
// MI455X (gfx1250) — compile-verified, matrix operand order checked by reading, not run on hardware
//
#include <hip/hip_runtime.h>
#include <hip/hip_bf16.h>

typedef __attribute__((ext_vector_type(16))) _Float16 v16h;
typedef __attribute__((ext_vector_type(8)))  _Float16 v8h;
typedef __attribute__((ext_vector_type(8)))  float    v8f;

#define C_      256
#define NH_     8
#define NL_     4
#define NP_     4
#define NLAYERS_ 6
#define D_      32
#define DFF_    1024
#define B_      2
#define S_      8500          // 80*80 + 40*40 + 20*20 + 10*10
#define NBS_    (B_ * S_)     // 17000 tokens
#define MPAD_   17008         // NBS_ rounded up to 16

// ---------------------------------------------------------------------------
// Weight transpose + f32 -> f16 convert:  W[K,N] (row major) -> Wt[N,K] f16
// ---------------------------------------------------------------------------
__global__ void convert_weight_t(const float* __restrict__ W,
                                 _Float16* __restrict__ Wt,
                                 int K, int N) {
    int idx = blockIdx.x * blockDim.x + threadIdx.x;
    if (idx >= K * N) return;
    int n = idx / K;
    int k = idx % K;
    Wt[idx] = (_Float16)W[(size_t)k * N + n];
}

// Zero a small f16 region (pad rows of the activation buffers)
__global__ void zero_f16(_Float16* __restrict__ p, int n) {
    int i = blockIdx.x * blockDim.x + threadIdx.x;
    if (i < n) p[i] = (_Float16)0.0f;
}

// out = src (f32) and out16 = f16(src)
__global__ void init_out(const float* __restrict__ src,
                         float* __restrict__ outF,
                         _Float16* __restrict__ outH, int n) {
    int i = blockIdx.x * blockDim.x + threadIdx.x;
    if (i >= n) return;
    float v = src[i];
    outF[i] = v;
    outH[i] = (_Float16)v;
}

// q16 = f16(out + pos)
__global__ void add_pos_f16(const float* __restrict__ out,
                            const float* __restrict__ pos,
                            _Float16* __restrict__ q16, int n) {
    int i = blockIdx.x * blockDim.x + threadIdx.x;
    if (i >= n) return;
    q16[i] = (_Float16)(out[i] + pos[i]);
}

// ---------------------------------------------------------------------------
// WMMA GEMM:  out[M,N] = X[Mpad,K] @ Wt[N,K]^T + bias, optional ReLU.
// One wave computes a 16x32 output slab (two 16x16 tiles sharing the A
// fragment) with v_wmma_f32_16x16x32_f16.  8 waves/block -> 256 columns.
// X must be padded to a multiple of 16 rows (pad rows zeroed) so A loads
// are unconditional -> no EXEC manipulation in the inner loop.
// ---------------------------------------------------------------------------
__global__ __launch_bounds__(256) void gemm_wmma_f16(
    const _Float16* __restrict__ X,    // [Mpad,K] f16, pad rows zero
    const _Float16* __restrict__ Wt,   // [N,K] f16 (W transposed)
    const float*    __restrict__ bias, // [N]
    float*          __restrict__ outF, // [M,N] f32 (or nullptr)
    _Float16*       __restrict__ outH, // [M,N] f16 (or nullptr)
    int M, int N, int K, int relu)
{
    const int lane = threadIdx.x & 31;
    const int wave = threadIdx.x >> 5;
    const int m0   = blockIdx.x * 16;
    const int n0   = (blockIdx.y * 8 + wave) * 32;   // 2 n-tiles per wave
    if (n0 >= N) return;                              // wave-uniform exit

    const int mr = lane & 15;   // row (A) / col (B,C) within tile
    const int h  = lane >> 4;   // lane half selects K octet

    const _Float16* xrow  = X  + (size_t)(m0 + mr) * K + 8 * h;
    const _Float16* wrow0 = Wt + (size_t)(n0 + mr) * K + 8 * h;
    const _Float16* wrow1 = wrow0 + (size_t)16 * K;

    v8f acc0 = {};
    v8f acc1 = {};
    for (int k0 = 0; k0 < K; k0 += 32) {
        v8h a0 = *(const v8h*)(xrow + k0);
        v8h a1 = *(const v8h*)(xrow + k0 + 16);
        v8h b00 = *(const v8h*)(wrow0 + k0);
        v8h b01 = *(const v8h*)(wrow0 + k0 + 16);
        v8h b10 = *(const v8h*)(wrow1 + k0);
        v8h b11 = *(const v8h*)(wrow1 + k0 + 16);
        v16h a  = __builtin_shufflevector(a0,  a1,  0,1,2,3,4,5,6,7,8,9,10,11,12,13,14,15);
        v16h bA = __builtin_shufflevector(b00, b01, 0,1,2,3,4,5,6,7,8,9,10,11,12,13,14,15);
        v16h bB = __builtin_shufflevector(b10, b11, 0,1,2,3,4,5,6,7,8,9,10,11,12,13,14,15);
        acc0 = __builtin_amdgcn_wmma_f32_16x16x32_f16(
            false, a, false, bA, (short)0, acc0, false, false);
        acc1 = __builtin_amdgcn_wmma_f32_16x16x32_f16(
            false, a, false, bB, (short)0, acc1, false, false);
    }

    const float bv0 = bias ? bias[n0 + mr]      : 0.0f;
    const float bv1 = bias ? bias[n0 + 16 + mr] : 0.0f;
    #pragma unroll
    for (int r = 0; r < 8; ++r) {
        int mm = m0 + r + 8 * h;             // C/D layout: row = r + 8*h
        if (mm < M) {
            size_t o = (size_t)mm * N + n0 + mr;
            float v0 = acc0[r] + bv0;
            float v1 = acc1[r] + bv1;
            if (relu) { v0 = fmaxf(v0, 0.0f); v1 = fmaxf(v1, 0.0f); }
            if (outF) { outF[o] = v0; outF[o + 16] = v1; }
            if (outH) { outH[o] = (_Float16)v0; outH[o + 16] = (_Float16)v1; }
        }
    }
}

// ---------------------------------------------------------------------------
// Softmax over 16 contiguous values per (b,s,head)
// ---------------------------------------------------------------------------
__global__ void softmax16(float* __restrict__ aw, int groups) {
    int g = blockIdx.x * blockDim.x + threadIdx.x;
    if (g >= groups) return;
    float* p = aw + (size_t)g * 16;
    float mx = p[0];
    #pragma unroll
    for (int i = 1; i < 16; ++i) mx = fmaxf(mx, p[i]);
    float s = 0.0f;
    float e[16];
    #pragma unroll
    for (int i = 0; i < 16; ++i) { e[i] = __expf(p[i] - mx); s += e[i]; }
    float inv = 1.0f / s;
    #pragma unroll
    for (int i = 0; i < 16; ++i) p[i] = e[i] * inv;
}

// ---------------------------------------------------------------------------
// Multi-scale deformable attention sampling.
// One wave per (b, q, head); lane = channel d (coalesced value reads).
// ---------------------------------------------------------------------------
__device__ __forceinline__ float samp_corner(const float* __restrict__ vb,
                                             int xi, int yi, int Wl, int Hl) {
    bool valid = (xi >= 0) && (xi < Wl) && (yi >= 0) && (yi < Hl);
    int cx = min(max(xi, 0), Wl - 1);
    int cy = min(max(yi, 0), Hl - 1);
    float v = vb[(size_t)(cy * Wl + cx) * (NH_ * D_)];
    return valid ? v : 0.0f;
}

__global__ __launch_bounds__(256) void deform_attn(
    const float* __restrict__ value,  // [B,S,NH,D]
    const float* __restrict__ off,    // [B,S,NH,NL,NP,2]
    const float* __restrict__ aw,     // [B,S,NH,NL,NP]
    const float* __restrict__ vr,     // [B,NL,2]
    _Float16*    __restrict__ attn16) // [B,S,NH,D]
{
    const int lane = threadIdx.x & 31;                 // d
    int w = blockIdx.x * 8 + (threadIdx.x >> 5);
    if (w >= B_ * S_ * NH_) return;
    const int hh = w % NH_;
    const int q  = (w / NH_) % S_;
    const int b  = w / (NH_ * S_);

    const int Hs[4] = {80, 40, 20, 10};
    const int Ws[4] = {80, 40, 20, 10};
    const int st[4] = {0, 6400, 8000, 8400};

    const int lq  = (q < 6400) ? 0 : (q < 8000) ? 1 : (q < 8400) ? 2 : 3;
    const int qi  = q - st[lq];
    const int row = qi / Ws[lq];
    const int col = qi % Ws[lq];
    const float bx = (col + 0.5f) / (vr[(b * NL_ + lq) * 2 + 0] * (float)Ws[lq]);
    const float by = (row + 0.5f) / (vr[(b * NL_ + lq) * 2 + 1] * (float)Hs[lq]);

    const float* offp = off + (size_t)((b * S_ + q) * NH_ + hh) * (NL_ * NP_ * 2);
    const float* awp  = aw  + (size_t)((b * S_ + q) * NH_ + hh) * (NL_ * NP_);

    float acc = 0.0f;
    #pragma unroll
    for (int lvl = 0; lvl < NL_; ++lvl) {
        const int Hl = Hs[lvl], Wl = Ws[lvl];
        const float rx = bx * vr[(b * NL_ + lvl) * 2 + 0];
        const float ry = by * vr[(b * NL_ + lvl) * 2 + 1];
        const float* vbase =
            value + ((size_t)b * S_ + st[lvl]) * (NH_ * D_) + hh * D_ + lane;
        #pragma unroll
        for (int p = 0; p < NP_; ++p) {
            float ox = offp[(lvl * NP_ + p) * 2 + 0];
            float oy = offp[(lvl * NP_ + p) * 2 + 1];
            float x = (rx + ox / (float)Wl) * (float)Wl - 0.5f;
            float y = (ry + oy / (float)Hl) * (float)Hl - 0.5f;
            float xf = floorf(x), yf = floorf(y);
            int x0 = (int)xf, y0 = (int)yf;
            int x1 = x0 + 1,  y1 = y0 + 1;
            float wx1 = x - xf, wx0 = 1.0f - wx1;
            float wy1 = y - yf, wy0 = 1.0f - wy1;
            float a = awp[lvl * NP_ + p];
            acc += a * (wx0 * wy0 * samp_corner(vbase, x0, y0, Wl, Hl) +
                        wx1 * wy0 * samp_corner(vbase, x1, y0, Wl, Hl) +
                        wx0 * wy1 * samp_corner(vbase, x0, y1, Wl, Hl) +
                        wx1 * wy1 * samp_corner(vbase, x1, y1, Wl, Hl));
        }
    }
    attn16[(size_t)((b * S_ + q) * NH_ + hh) * D_ + lane] = (_Float16)acc;
}

// ---------------------------------------------------------------------------
// out = LayerNorm(base + delta) * g + b; writes f32 (+ optional f16 copy)
// One block per token; blockDim.x == C_ == 256.
// ---------------------------------------------------------------------------
__global__ __launch_bounds__(256) void residual_ln(
    const float* __restrict__ base, const float* __restrict__ delta,
    const float* __restrict__ g,    const float* __restrict__ bb,
    float* __restrict__ outF, _Float16* __restrict__ outH)
{
    __shared__ float red[C_];
    const int t = blockIdx.x;
    const int c = threadIdx.x;
    const size_t o = (size_t)t * C_ + c;

    float x = base[o] + delta[o];

    red[c] = x;
    __syncthreads();
    for (int s = C_ / 2; s > 0; s >>= 1) {
        if (c < s) red[c] += red[c + s];
        __syncthreads();
    }
    const float mean = red[0] * (1.0f / C_);
    __syncthreads();

    const float d = x - mean;
    red[c] = d * d;
    __syncthreads();
    for (int s = C_ / 2; s > 0; s >>= 1) {
        if (c < s) red[c] += red[c + s];
        __syncthreads();
    }
    const float var = red[0] * (1.0f / C_);

    const float y = d * rsqrtf(var + 1e-5f) * g[c] + bb[c];
    outF[o] = y;
    if (outH) outH[o] = (_Float16)y;
}

// ---------------------------------------------------------------------------
// Host-side orchestration
// ---------------------------------------------------------------------------
static inline char* ws_take(char*& p, size_t bytes) {
    char* r = p;
    p += (bytes + 255) & ~(size_t)255;
    return r;
}

extern "C" void kernel_launch(void* const* d_in, const int* in_sizes, int n_in,
                              void* d_out, int out_size, void* d_ws, size_t ws_size,
                              hipStream_t stream) {
    (void)in_sizes; (void)n_in; (void)out_size; (void)ws_size;

    const float* src     = (const float*)d_in[0];
    const float* pos     = (const float*)d_in[1];
    const float* vr      = (const float*)d_in[2];
    // d_in[3] = spatial_shapes (int32), baked in as constants
    const float* off_w   = (const float*)d_in[4];
    const float* off_b   = (const float*)d_in[5];
    const float* aw_w    = (const float*)d_in[6];
    const float* aw_b    = (const float*)d_in[7];
    const float* vproj_w = (const float*)d_in[8];
    const float* vproj_b = (const float*)d_in[9];
    const float* oproj_w = (const float*)d_in[10];
    const float* oproj_b = (const float*)d_in[11];
    const float* ln1_g   = (const float*)d_in[12];
    const float* ln1_b   = (const float*)d_in[13];
    const float* ffn1_w  = (const float*)d_in[14];
    const float* ffn1_b  = (const float*)d_in[15];
    const float* ffn2_w  = (const float*)d_in[16];
    const float* ffn2_b  = (const float*)d_in[17];
    const float* ln2_g   = (const float*)d_in[18];
    const float* ln2_b   = (const float*)d_in[19];
    float* final_out = (float*)d_out;

    const int M    = NBS_;             // 17000
    const int Mpad = MPAD_;            // 17008
    const int NE   = NBS_ * C_;        // elementwise count (B*S*C)
    const int PAD  = Mpad - M;         // 8 pad rows

    // ---- workspace carve-up (f16 activations padded to Mpad rows) ----
    char* p = (char*)d_ws;
    float*    outF     = (float*)   ws_take(p, (size_t)NE * 4);
    float*    valueF   = (float*)   ws_take(p, (size_t)NE * 4);
    float*    offF     = (float*)   ws_take(p, (size_t)NE * 4);
    float*    attnOutF = (float*)   ws_take(p, (size_t)NE * 4);
    float*    ffnOutF  = (float*)   ws_take(p, (size_t)NE * 4);
    float*    awF      = (float*)   ws_take(p, (size_t)M * 128 * 4);
    _Float16* out16    = (_Float16*)ws_take(p, (size_t)Mpad * C_ * 2);
    _Float16* q16      = (_Float16*)ws_take(p, (size_t)Mpad * C_ * 2);
    _Float16* attn16   = (_Float16*)ws_take(p, (size_t)Mpad * C_ * 2);
    _Float16* h16      = (_Float16*)ws_take(p, (size_t)Mpad * DFF_ * 2);
    _Float16* vprojT   = (_Float16*)ws_take(p, (size_t)NLAYERS_ * C_ * C_ * 2);
    _Float16* offT     = (_Float16*)ws_take(p, (size_t)NLAYERS_ * C_ * C_ * 2);
    _Float16* awT      = (_Float16*)ws_take(p, (size_t)NLAYERS_ * C_ * 128 * 2);
    _Float16* oprojT   = (_Float16*)ws_take(p, (size_t)NLAYERS_ * C_ * C_ * 2);
    _Float16* ffn1T    = (_Float16*)ws_take(p, (size_t)NLAYERS_ * C_ * DFF_ * 2);
    _Float16* ffn2T    = (_Float16*)ws_take(p, (size_t)NLAYERS_ * DFF_ * C_ * 2);

    // ---- zero the pad rows of every f16 activation buffer ----
    zero_f16<<<(PAD * C_   + 255)/256, 256, 0, stream>>>(out16  + (size_t)M * C_,   PAD * C_);
    zero_f16<<<(PAD * C_   + 255)/256, 256, 0, stream>>>(q16    + (size_t)M * C_,   PAD * C_);
    zero_f16<<<(PAD * C_   + 255)/256, 256, 0, stream>>>(attn16 + (size_t)M * C_,   PAD * C_);
    zero_f16<<<(PAD * DFF_ + 255)/256, 256, 0, stream>>>(h16    + (size_t)M * DFF_, PAD * DFF_);

    // ---- one-time weight transpose+convert (graph-safe, deterministic) ----
    for (int i = 0; i < NLAYERS_; ++i) {
        {   int K = C_, N = C_;
            convert_weight_t<<<(K*N + 255)/256, 256, 0, stream>>>(
                vproj_w + (size_t)i*K*N, vprojT + (size_t)i*N*K, K, N); }
        {   int K = C_, N = C_;
            convert_weight_t<<<(K*N + 255)/256, 256, 0, stream>>>(
                off_w + (size_t)i*K*N, offT + (size_t)i*N*K, K, N); }
        {   int K = C_, N = 128;
            convert_weight_t<<<(K*N + 255)/256, 256, 0, stream>>>(
                aw_w + (size_t)i*K*N, awT + (size_t)i*N*K, K, N); }
        {   int K = C_, N = C_;
            convert_weight_t<<<(K*N + 255)/256, 256, 0, stream>>>(
                oproj_w + (size_t)i*K*N, oprojT + (size_t)i*N*K, K, N); }
        {   int K = C_, N = DFF_;
            convert_weight_t<<<(K*N + 255)/256, 256, 0, stream>>>(
                ffn1_w + (size_t)i*K*N, ffn1T + (size_t)i*N*K, K, N); }
        {   int K = DFF_, N = C_;
            convert_weight_t<<<(K*N + 255)/256, 256, 0, stream>>>(
                ffn2_w + (size_t)i*K*N, ffn2T + (size_t)i*N*K, K, N); }
    }

    // ---- out = src ----
    init_out<<<(NE + 255)/256, 256, 0, stream>>>(src, outF, out16, NE);

    const int Mtiles = Mpad / 16;       // 1063
    const dim3 blk(256);
    auto gy = [](int N) { return (unsigned)((N + 255) / 256); };

    for (int i = 0; i < NLAYERS_; ++i) {
        // q16 = f16(out + pos)
        add_pos_f16<<<(NE + 255)/256, 256, 0, stream>>>(outF, pos, q16, NE);

        // value = out @ vproj + b
        gemm_wmma_f16<<<dim3(Mtiles, gy(C_)), blk, 0, stream>>>(
            out16, vprojT + (size_t)i*C_*C_, vproj_b + (size_t)i*C_,
            valueF, (_Float16*)nullptr, M, C_, C_, 0);

        // off = q @ off_w + b
        gemm_wmma_f16<<<dim3(Mtiles, gy(C_)), blk, 0, stream>>>(
            q16, offT + (size_t)i*C_*C_, off_b + (size_t)i*C_,
            offF, (_Float16*)nullptr, M, C_, C_, 0);

        // aw = softmax(q @ aw_w + b)
        gemm_wmma_f16<<<dim3(Mtiles, gy(128)), blk, 0, stream>>>(
            q16, awT + (size_t)i*C_*128, aw_b + (size_t)i*128,
            awF, (_Float16*)nullptr, M, 128, C_, 0);
        softmax16<<<(M*NH_ + 255)/256, 256, 0, stream>>>(awF, M * NH_);

        // multi-scale deformable sampling -> attn16
        deform_attn<<<(M*NH_ + 7)/8, blk, 0, stream>>>(
            valueF, offF, awF, vr, attn16);

        // attn_out = attn @ oproj + b
        gemm_wmma_f16<<<dim3(Mtiles, gy(C_)), blk, 0, stream>>>(
            attn16, oprojT + (size_t)i*C_*C_, oproj_b + (size_t)i*C_,
            attnOutF, (_Float16*)nullptr, M, C_, C_, 0);

        // out = LN(out + attn_out)
        residual_ln<<<M, C_, 0, stream>>>(
            outF, attnOutF, ln1_g + (size_t)i*C_, ln1_b + (size_t)i*C_,
            outF, out16);

        // h = relu(out @ ffn1 + b)  (f16 only)
        gemm_wmma_f16<<<dim3(Mtiles, gy(DFF_)), blk, 0, stream>>>(
            out16, ffn1T + (size_t)i*C_*DFF_, ffn1_b + (size_t)i*DFF_,
            (float*)nullptr, h16, M, DFF_, C_, 1);

        // ffn_out = h @ ffn2 + b
        gemm_wmma_f16<<<dim3(Mtiles, gy(C_)), blk, 0, stream>>>(
            h16, ffn2T + (size_t)i*DFF_*C_, ffn2_b + (size_t)i*C_,
            ffnOutF, (_Float16*)nullptr, M, C_, DFF_, 0);

        // out = LN(out + ffn_out); last layer writes directly to d_out
        float* dst = (i == NLAYERS_ - 1) ? final_out : outF;
        residual_ln<<<M, C_, 0, stream>>>(
            outF, ffnOutF, ln2_g + (size_t)i*C_, ln2_b + (size_t)i*C_,
            dst, out16);
    }
}
